// TwoStreamCABlock_29472065585428
// MI455X (gfx1250) — matrix operand order repaired and checked by reading, not compiled, not run on hardware
//
#include <hip/hip_runtime.h>
#include <hip/hip_bf16.h>

// ---------------------------------------------------------------------------
// TwoStreamCABlock for MI455X (gfx1250): all GEMMs + attention via
// v_wmma_f32_16x16x32_f16 (f16 inputs, f32 accumulate). Software-pipelined
// LDS staging, uniform fast paths to avoid EXEC-masked serialized loads.
// ---------------------------------------------------------------------------

typedef __attribute__((ext_vector_type(4)))  _Float16 v4h;
typedef __attribute__((ext_vector_type(8)))  _Float16 v8h;
typedef __attribute__((ext_vector_type(16))) _Float16 v16h;
typedef __attribute__((ext_vector_type(4)))  float    v4f;
typedef __attribute__((ext_vector_type(8)))  float    v8f;

#define HIDN   1536
#define NHEAD  12
#define HD     128
#define SEQA   2048          // tokens per stream
#define SEQT   4096          // total (audio + cond)
#define LTXT   100
#define MLPH   6144
#define EPSN   1e-6f

__device__ __forceinline__ v16h cat8(v8h lo, v8h hi) {
  return __builtin_shufflevector(lo, hi, 0,1,2,3,4,5,6,7,8,9,10,11,12,13,14,15);
}
__device__ __forceinline__ v8f wmma16(v16h a, v16h b, v8f c) {
  // D = A(16x32 f16) * B(32x16 f16) + C(16x16 f32)
  return __builtin_amdgcn_wmma_f32_16x16x32_f16(false, a, false, b, (short)0, c,
                                                false, false);
}
__device__ __forceinline__ v8f zero8f() {
  v8f z = {0.f,0.f,0.f,0.f,0.f,0.f,0.f,0.f};
  return z;
}
__device__ __forceinline__ v8h zero8h() {
  v8h z = {(_Float16)0,(_Float16)0,(_Float16)0,(_Float16)0,
           (_Float16)0,(_Float16)0,(_Float16)0,(_Float16)0};
  return z;
}
__device__ __forceinline__ float gelu_tanh(float x) {
  float t = tanhf(0.7978845608028654f * (x + 0.044715f * x * x * x));
  return 0.5f * x * (1.f + t);
}

// ---------------------------------------------------------------------------
// Kernel: s = silu(vec)
// ---------------------------------------------------------------------------
__global__ __launch_bounds__(256) void silu_vec_kernel(const float* __restrict__ vec,
                                                       float* __restrict__ s) {
  int i = blockIdx.x * 256 + threadIdx.x;
  if (i < HIDN) {
    float x = vec[i];
    s[i] = x / (1.f + __expf(-x));
  }
}

// ---------------------------------------------------------------------------
// Kernel: mod vectors  out[j] = sum_k s[k]*W[k, j] + b[j]  for both streams
// ---------------------------------------------------------------------------
__global__ __launch_bounds__(256) void modvec_kernel(const float* __restrict__ s,
                                                     const float* __restrict__ aw,
                                                     const float* __restrict__ ab,
                                                     const float* __restrict__ cw,
                                                     const float* __restrict__ cb,
                                                     float* __restrict__ amod,
                                                     float* __restrict__ cmod) {
  int j = blockIdx.x * 256 + threadIdx.x;  // 0 .. 2*13824-1
  const int NW = 9 * HIDN;
  int isA = (j < NW) ? 1 : 0;
  int jj = isA ? j : j - NW;
  const float* W = isA ? aw : cw;
  const float* B = isA ? ab : cb;
  float acc = 0.f;
  for (int k = 0; k < HIDN; ++k) acc += s[k] * W[(size_t)k * NW + jj];
  float* out = isA ? amod : cmod;
  out[jj] = acc + B[jj];
}

// ---------------------------------------------------------------------------
// Kernel: layer_norm(x) * (1+scale) + shift   (one block per row of 1536)
// ---------------------------------------------------------------------------
__global__ __launch_bounds__(256) void ln_mod_kernel(const float* __restrict__ x,
                                                     const float* __restrict__ shift,
                                                     const float* __restrict__ scale,
                                                     float* __restrict__ out) {
  __shared__ float red[8];
  const int row = blockIdx.x, tid = threadIdx.x;
  const float* xr = x + (size_t)row * HIDN;
  float v[6];
  float sum = 0.f;
#pragma unroll
  for (int i = 0; i < 6; ++i) { v[i] = xr[tid + 256 * i]; sum += v[i]; }
#pragma unroll
  for (int o = 1; o < 32; o <<= 1) sum += __shfl_xor(sum, o, 32);
  if ((tid & 31) == 0) red[tid >> 5] = sum;
  __syncthreads();
  float tot = 0.f;
#pragma unroll
  for (int i = 0; i < 8; ++i) tot += red[i];
  float mu = tot * (1.f / HIDN);
  __syncthreads();
  float s2 = 0.f;
#pragma unroll
  for (int i = 0; i < 6; ++i) { float d = v[i] - mu; s2 += d * d; }
#pragma unroll
  for (int o = 1; o < 32; o <<= 1) s2 += __shfl_xor(s2, o, 32);
  if ((tid & 31) == 0) red[tid >> 5] = s2;
  __syncthreads();
  float tot2 = 0.f;
#pragma unroll
  for (int i = 0; i < 8; ++i) tot2 += red[i];
  float rstd = rsqrtf(tot2 * (1.f / HIDN) + EPSN);
#pragma unroll
  for (int i = 0; i < 6; ++i) {
    int c = tid + 256 * i;
    out[(size_t)row * HIDN + c] = (v[i] - mu) * rstd * (1.f + scale[c]) + shift[c];
  }
}

// ---------------------------------------------------------------------------
// Generic WMMA GEMM: C(MxN,f32) = A(MxK,f32) * B(KxN,f32) with fused epilogue.
// Block: 256 thr / 8 waves, tile BM=128 BN=64 BK=64, wave = 2x2 16x16 tiles.
// Software-pipelined: next tile loaded to registers during WMMA of current.
// ---------------------------------------------------------------------------
#define EPI_NONE       0
#define EPI_BIAS_GELU  1
#define EPI_GATE_RES   2

__global__ __launch_bounds__(256)
void gemm_kernel(const float* __restrict__ A, const float* __restrict__ B,
                 float* __restrict__ C, int M, int N, int K, int epi,
                 const float* __restrict__ bias, const float* __restrict__ gate,
                 const float* __restrict__ res) {
  __shared__ __align__(16) _Float16 sA[128 * 64];   // [m][k]
  __shared__ __align__(16) _Float16 sB[64 * 64];    // transposed: [n][k]
  const int tid = threadIdx.x;
  const int wave = tid >> 5, lane = tid & 31;
  const int hl = lane & 15, lh = lane >> 4;
  const int wm = wave >> 1, wn = wave & 1;
  const int m0 = blockIdx.x * 128;
  const int n0 = blockIdx.y * 64;
  const bool fullM = (m0 + 128 <= M);   // block-uniform fast path

  v4f ra[8], rb[4];

  auto load_tiles = [&](int k0) {
    // B panel (64x64) as float4, always in-bounds
#pragma unroll
    for (int i = 0; i < 4; ++i) {
      int c = tid + 256 * i;
      int k = c >> 4, nq = (c & 15) * 4;
      rb[i] = *(const v4f*)(B + (size_t)(k0 + k) * N + n0 + nq);
    }
    // prefetch the following B panel into caches (global_prefetch_b8)
    if (k0 + 128 <= K)
      __builtin_prefetch((const void*)(B + (size_t)(k0 + 64 + (tid >> 2)) * N +
                                       n0 + (tid & 3) * 16), 0, 3);
    // A panel (128x64) as float4
    if (fullM) {
#pragma unroll
      for (int i = 0; i < 8; ++i) {
        int c = tid + 256 * i;
        int m = c >> 4, kq = (c & 15) * 4;
        ra[i] = *(const v4f*)(A + (size_t)(m0 + m) * K + k0 + kq);
      }
    } else {
#pragma unroll
      for (int i = 0; i < 8; ++i) {
        int c = tid + 256 * i;
        int m = c >> 4, kq = (c & 15) * 4;
        v4f z = {0.f, 0.f, 0.f, 0.f};
        ra[i] = (m0 + m < M) ? *(const v4f*)(A + (size_t)(m0 + m) * K + k0 + kq)
                             : z;
      }
    }
  };

  v8f acc[2][2];
#pragma unroll
  for (int a = 0; a < 2; ++a)
#pragma unroll
    for (int b = 0; b < 2; ++b) acc[a][b] = zero8f();

  load_tiles(0);
  for (int k0 = 0; k0 < K; k0 += 64) {
    __syncthreads();   // prior compute finished reading LDS
    // store staged registers to LDS (f32 -> f16)
#pragma unroll
    for (int i = 0; i < 8; ++i) {
      int c = tid + 256 * i;
      int m = c >> 4, kq = (c & 15) * 4;
      v4h hh;
      hh[0] = (_Float16)ra[i][0]; hh[1] = (_Float16)ra[i][1];
      hh[2] = (_Float16)ra[i][2]; hh[3] = (_Float16)ra[i][3];
      *(v4h*)&sA[m * 64 + kq] = hh;
    }
#pragma unroll
    for (int i = 0; i < 4; ++i) {
      int c = tid + 256 * i;
      int k = c >> 4, nq = (c & 15) * 4;
#pragma unroll
      for (int j = 0; j < 4; ++j) sB[(nq + j) * 64 + k] = (_Float16)rb[i][j];
    }
    __syncthreads();
    if (k0 + 64 < K) load_tiles(k0 + 64);   // hide global latency behind WMMA
#pragma unroll
    for (int kk = 0; kk < 64; kk += 32) {
      v16h af[2], bf[2];
#pragma unroll
      for (int tm = 0; tm < 2; ++tm) {
        int r = wm * 32 + tm * 16 + hl;
        v8h lo = *(const v8h*)&sA[r * 64 + kk + lh * 16];
        v8h hi = *(const v8h*)&sA[r * 64 + kk + lh * 16 + 8];
        af[tm] = cat8(lo, hi);
      }
#pragma unroll
      for (int tn = 0; tn < 2; ++tn) {
        int c = wn * 32 + tn * 16 + hl;
        bf[tn] = *(const v16h*)&sB[c * 64 + kk + lh * 16];
      }
#pragma unroll
      for (int tm = 0; tm < 2; ++tm)
#pragma unroll
        for (int tn = 0; tn < 2; ++tn)
          acc[tm][tn] = wmma16(af[tm], bf[tn], acc[tm][tn]);
    }
  }

#pragma unroll
  for (int tm = 0; tm < 2; ++tm)
#pragma unroll
    for (int tn = 0; tn < 2; ++tn)
#pragma unroll
      for (int j = 0; j < 8; ++j) {
        int gm = m0 + wm * 32 + tm * 16 + j + (lh << 3);
        int gn = n0 + wn * 32 + tn * 16 + hl;
        if (gm < M) {
          float v = acc[tm][tn][j];
          if (epi == EPI_BIAS_GELU) {
            v = gelu_tanh(v + bias[gn]);
          } else if (epi == EPI_GATE_RES) {
            if (bias) v += bias[gn];
            v = res[(size_t)gm * N + gn] + v * gate[gn];
          }
          C[(size_t)gm * N + gn] = v;
        }
      }
}

// ---------------------------------------------------------------------------
// Kernel: qkv post-process. One wave per (token, head):
//   q,k: rms_norm(128, w) then RoPE with freq row = 2*token + stream; v: copy.
//   Writes f16 at rows [stream*2048 + token] (== interleave+decouple+concat).
// ---------------------------------------------------------------------------
__global__ __launch_bounds__(256)
void qkv_post_kernel(const float* __restrict__ qkv, const float* __restrict__ qw,
                     const float* __restrict__ kw, const float* __restrict__ fcos,
                     const float* __restrict__ fsin, _Float16* __restrict__ qh,
                     _Float16* __restrict__ kh, _Float16* __restrict__ vh,
                     int stream_id) {
  const int wave = threadIdx.x >> 5, lane = threadIdx.x & 31;
  const int gid = blockIdx.x * 8 + wave;
  const int tok = gid / NHEAD, head = gid % NHEAD;
  if (tok >= SEQA) return;
  const int d0 = lane * 4;
  const float* base = qkv + (size_t)tok * (3 * HIDN) + head * HD;
  const size_t orow = (size_t)(stream_id * SEQA + tok) * HIDN + head * HD + d0;
  const size_t frow = (size_t)(2 * tok + stream_id) * HD + d0;
  float c0 = fcos[frow], c1 = fcos[frow + 1], c2 = fcos[frow + 2], c3 = fcos[frow + 3];
  float s0 = fsin[frow], s1 = fsin[frow + 1], s2 = fsin[frow + 2], s3 = fsin[frow + 3];

#pragma unroll
  for (int which = 0; which < 2; ++which) {
    const float* p = base + which * HIDN;
    const float* w = which ? kw : qw;
    float x0 = p[d0], x1 = p[d0 + 1], x2 = p[d0 + 2], x3 = p[d0 + 3];
    float ss = x0 * x0 + x1 * x1 + x2 * x2 + x3 * x3;
#pragma unroll
    for (int o = 1; o < 32; o <<= 1) ss += __shfl_xor(ss, o, 32);
    float r = rsqrtf(ss * (1.f / HD) + EPSN);
    float y0 = x0 * r * w[d0],     y1 = x1 * r * w[d0 + 1];
    float y2 = x2 * r * w[d0 + 2], y3 = x3 * r * w[d0 + 3];
    float o0 = y0 * c0 - y1 * s0;
    float o1 = y1 * c1 + y0 * s1;
    float o2 = y2 * c2 - y3 * s2;
    float o3 = y3 * c3 + y2 * s3;
    _Float16* out = which ? kh : qh;
    out[orow] = (_Float16)o0; out[orow + 1] = (_Float16)o1;
    out[orow + 2] = (_Float16)o2; out[orow + 3] = (_Float16)o3;
  }
  const float* pv = base + 2 * HIDN;
  vh[orow]     = (_Float16)pv[d0];
  vh[orow + 1] = (_Float16)pv[d0 + 1];
  vh[orow + 2] = (_Float16)pv[d0 + 2];
  vh[orow + 3] = (_Float16)pv[d0 + 3];
}

// ---------------------------------------------------------------------------
// Kernel: per-head rms_norm (or plain convert if w==null) f32 -> f16.
// One wave per (token, head). inStride = floats per token row in `in`.
// ---------------------------------------------------------------------------
__global__ __launch_bounds__(256)
void rms_head_kernel(const float* __restrict__ in, int inStride, int inBase,
                     const float* __restrict__ w, _Float16* __restrict__ out,
                     int nTok) {
  const int wave = threadIdx.x >> 5, lane = threadIdx.x & 31;
  const int gid = blockIdx.x * 8 + wave;
  const int tok = gid / NHEAD, head = gid % NHEAD;
  if (tok >= nTok) return;
  const int d0 = lane * 4;
  const float* p = in + (size_t)tok * inStride + inBase + head * HD;
  float x0 = p[d0], x1 = p[d0 + 1], x2 = p[d0 + 2], x3 = p[d0 + 3];
  if (w) {
    float ss = x0 * x0 + x1 * x1 + x2 * x2 + x3 * x3;
#pragma unroll
    for (int o = 1; o < 32; o <<= 1) ss += __shfl_xor(ss, o, 32);
    float r = rsqrtf(ss * (1.f / HD) + EPSN);
    x0 *= r * w[d0]; x1 *= r * w[d0 + 1]; x2 *= r * w[d0 + 2]; x3 *= r * w[d0 + 3];
  }
  size_t orow = (size_t)tok * HIDN + head * HD + d0;
  out[orow] = (_Float16)x0; out[orow + 1] = (_Float16)x1;
  out[orow + 2] = (_Float16)x2; out[orow + 3] = (_Float16)x3;
}

// ---------------------------------------------------------------------------
// Flash attention (WMMA). Grid: (Lq/128, H). 8 waves x 16 q-rows.
// Q,K,V are f16 with row stride 1536 (H*D); O is f32, same layout.
// Online softmax; 32-key tiles staged in LDS (software pipelined);
// uniform fast path except the masked tail tile (cross-attn Lk=100).
// ---------------------------------------------------------------------------
__global__ __launch_bounds__(256)
void flash_attn_kernel(const _Float16* __restrict__ Q, const _Float16* __restrict__ Km,
                       const _Float16* __restrict__ V, float* __restrict__ O,
                       int Lk, float scale) {
  __shared__ __align__(16) _Float16 sK[32 * 128];        // [key][d]
  __shared__ __align__(16) _Float16 sVT[128 * 32];       // [d][key]
  __shared__ __align__(16) _Float16 sP[8][16 * 32];      // per-wave P tile
  const int tid = threadIdx.x;
  const int wave = tid >> 5, lane = tid & 31;
  const int hl = lane & 15, lh = lane >> 4;
  const int h = blockIdx.y;
  const int qbase = blockIdx.x * 128 + wave * 16;

  // Q fragments: 4 x (16x32) covering d = 0..127
  const _Float16* qp = Q + (size_t)(qbase + hl) * HIDN + h * HD + lh * 16;
  v16h qf[4];
#pragma unroll
  for (int f = 0; f < 4; ++f) {
    v8h lo = *(const v8h*)(qp + f * 32);
    v8h hi = *(const v8h*)(qp + f * 32 + 8);
    qf[f] = cat8(lo, hi);
  }

  v8f o[8];
  float m_i[8], l_i[8];
#pragma unroll
  for (int t = 0; t < 8; ++t) { o[t] = zero8f(); m_i[t] = -3e38f; l_i[t] = 0.f; }

  v8h rk[2], rv[2];
  auto load_kv = [&](int kt) {
    const bool full = (kt * 32 + 32 <= Lk);    // block-uniform
#pragma unroll
    for (int i = 0; i < 2; ++i) {
      int c = tid + 256 * i;                    // v8h chunk id, 512 total
      int key = c >> 4, dp = c & 15;
      size_t off = (size_t)(kt * 32 + key) * HIDN + h * HD + dp * 8;
      if (full) {
        rk[i] = *(const v8h*)(Km + off);
        rv[i] = *(const v8h*)(V + off);
      } else {
        bool ok = (kt * 32 + key) < Lk;
        rk[i] = ok ? *(const v8h*)(Km + off) : zero8h();
        rv[i] = ok ? *(const v8h*)(V + off) : zero8h();
      }
    }
  };

  const int nkt = (Lk + 31) / 32;
  load_kv(0);
  for (int kt = 0; kt < nkt; ++kt) {
    __syncthreads();
    // store staged K/V registers to LDS (V transposed)
#pragma unroll
    for (int i = 0; i < 2; ++i) {
      int c = tid + 256 * i;
      int key = c >> 4, dp = c & 15;
      *(v8h*)&sK[key * 128 + dp * 8] = rk[i];
#pragma unroll
      for (int j = 0; j < 8; ++j) sVT[(dp * 8 + j) * 32 + key] = rv[i][j];
    }
    __syncthreads();
    if (kt + 1 < nkt) load_kv(kt + 1);   // hide global latency behind WMMA

    // S = Q * K^T  (two 16x16 C tiles: keys 0-15 and 16-31 of this tile)
    v8f s0 = zero8f();
    v8f s1 = zero8f();
#pragma unroll
    for (int f = 0; f < 4; ++f) {
      v16h b0 = *(const v16h*)&sK[hl * 128 + f * 32 + lh * 16];
      s0 = wmma16(qf[f], b0, s0);
      v16h b1 = *(const v16h*)&sK[(hl + 16) * 128 + f * 32 + lh * 16];
      s1 = wmma16(qf[f], b1, s1);
    }

    const bool msk0 = (kt * 32 + hl) >= Lk;
    const bool msk1 = (kt * 32 + 16 + hl) >= Lk;
    float alpha[8];
#pragma unroll
    for (int j = 0; j < 8; ++j) {
      float a = msk0 ? -3e38f : s0[j] * scale;
      float b = msk1 ? -3e38f : s1[j] * scale;
      float mx = fmaxf(a, b);
#pragma unroll
      for (int off = 1; off < 16; off <<= 1) mx = fmaxf(mx, __shfl_xor(mx, off, 32));
      float mn = fmaxf(m_i[j], mx);
      float al = __expf(m_i[j] - mn);
      m_i[j] = mn; alpha[j] = al;
      float p0 = msk0 ? 0.f : __expf(a - mn);
      float p1 = msk1 ? 0.f : __expf(b - mn);
      float rs = p0 + p1;
#pragma unroll
      for (int off = 1; off < 16; off <<= 1) rs += __shfl_xor(rs, off, 32);
      l_i[j] = l_i[j] * al + rs;
      s0[j] = p0; s1[j] = p1;
    }
#pragma unroll
    for (int t = 0; t < 8; ++t)
#pragma unroll
      for (int j = 0; j < 8; ++j) o[t][j] *= alpha[j];

    // reshape P (C-layout f32) -> A-fragment via per-wave LDS bounce
    _Float16* pw = sP[wave];
#pragma unroll
    for (int j = 0; j < 8; ++j) {
      int r = j + (lh << 3);
      pw[r * 32 + hl] = (_Float16)s0[j];
      pw[r * 32 + 16 + hl] = (_Float16)s1[j];
    }
    v8h plo = *(const v8h*)&pw[hl * 32 + lh * 16];
    v8h phi = *(const v8h*)&pw[hl * 32 + lh * 16 + 8];
    v16h pf = cat8(plo, phi);

    // O += P * V
#pragma unroll
    for (int t = 0; t < 8; ++t) {
      v16h bv = *(const v16h*)&sVT[(t * 16 + hl) * 32 + lh * 16];
      o[t] = wmma16(pf, bv, o[t]);
    }
  }

#pragma unroll
  for (int t = 0; t < 8; ++t)
#pragma unroll
    for (int j = 0; j < 8; ++j) {
      int row = qbase + j + (lh << 3);
      O[(size_t)row * HIDN + h * HD + t * 16 + hl] = o[t][j] / l_i[j];
    }
}

// ---------------------------------------------------------------------------
// Host-side orchestration
// ---------------------------------------------------------------------------
enum {
  IN_AUDIO = 0, IN_COND, IN_VEC, IN_TEXT, IN_FCOS, IN_FSIN,
  P_AMOD_W, P_AMOD_B, P_CMOD_W, P_CMOD_B,
  P_AQKV_W, P_CQKV_W, P_ASELF_PROJ_W, P_CSELF_PROJ_W,
  P_ACROSS_Q_W, P_CCROSS_Q_W, P_TEXT_KV_W,
  P_ACROSS_PROJ_W, P_CCROSS_PROJ_W,
  P_AQ_NORM, P_AK_NORM, P_CQ_NORM, P_CK_NORM,
  P_ACROSS_QN, P_CCROSS_QN, P_TEXT_KN,
  P_AFC1_W, P_AFC1_B, P_AFC2_W, P_AFC2_B,
  P_CFC1_W, P_CFC1_B, P_CFC2_W, P_CFC2_B
};

static void launch_gemm(const float* A, const float* B, float* C, int M, int N,
                        int K, int epi, const float* bias, const float* gate,
                        const float* res, hipStream_t stream) {
  dim3 grid((M + 127) / 128, N / 64);
  gemm_kernel<<<grid, 256, 0, stream>>>(A, B, C, M, N, K, epi, bias, gate, res);
}

extern "C" void kernel_launch(void* const* d_in, const int* in_sizes, int n_in,
                              void* d_out, int out_size, void* d_ws, size_t ws_size,
                              hipStream_t stream) {
  (void)in_sizes; (void)n_in; (void)out_size; (void)ws_size;
  const float* f_audio = (const float*)d_in[IN_AUDIO];
  const float* f_cond  = (const float*)d_in[IN_COND];
  const float* f_vec   = (const float*)d_in[IN_VEC];
  const float* f_text  = (const float*)d_in[IN_TEXT];
  const float* f_cos   = (const float*)d_in[IN_FCOS];
  const float* f_sin   = (const float*)d_in[IN_FSIN];

  // workspace layout
  char* w = (char*)d_ws;
  auto alloc = [&](size_t bytes) -> void* {
    void* p = (void*)w;
    w += (bytes + 255) & ~(size_t)255;
    return p;
  };
  float*     sbuf  = (float*)alloc((size_t)HIDN * 4);
  float*     amod  = (float*)alloc((size_t)9 * HIDN * 4);
  float*     cmod  = (float*)alloc((size_t)9 * HIDN * 4);
  _Float16*  qh    = (_Float16*)alloc((size_t)SEQT * HIDN * 2);
  _Float16*  kh    = (_Float16*)alloc((size_t)SEQT * HIDN * 2);
  _Float16*  vh    = (_Float16*)alloc((size_t)SEQT * HIDN * 2);
  float*     xmod  = (float*)alloc((size_t)SEQA * HIDN * 4);
  float*     big   = (float*)alloc((size_t)SEQA * MLPH * 4);   // qkv f32 / mlp hidden
  float*     attno = (float*)alloc((size_t)SEQT * HIDN * 4);
  float*     tkv   = (float*)alloc((size_t)LTXT * 2 * HIDN * 4);
  _Float16*  tkh   = (_Float16*)alloc((size_t)LTXT * HIDN * 2);
  _Float16*  tvh   = (_Float16*)alloc((size_t)LTXT * HIDN * 2);
  _Float16*  qxh   = (_Float16*)alloc((size_t)SEQA * HIDN * 2);

  const float scale = 0.08838834764831845f;  // 1/sqrt(128)

  // per-stream parameter tables
  const float* x_in[2]   = {f_audio, f_cond};
  const float* mods[2]   = {amod, cmod};
  const float* qkvw[2]   = {(const float*)d_in[P_AQKV_W], (const float*)d_in[P_CQKV_W]};
  const float* qn[2]     = {(const float*)d_in[P_AQ_NORM], (const float*)d_in[P_CQ_NORM]};
  const float* kn[2]     = {(const float*)d_in[P_AK_NORM], (const float*)d_in[P_CK_NORM]};
  const float* sprojw[2] = {(const float*)d_in[P_ASELF_PROJ_W], (const float*)d_in[P_CSELF_PROJ_W]};
  const float* cqw[2]    = {(const float*)d_in[P_ACROSS_Q_W], (const float*)d_in[P_CCROSS_Q_W]};
  const float* cqn[2]    = {(const float*)d_in[P_ACROSS_QN], (const float*)d_in[P_CCROSS_QN]};
  const float* cprojw[2] = {(const float*)d_in[P_ACROSS_PROJ_W], (const float*)d_in[P_CCROSS_PROJ_W]};
  const float* fc1w[2]   = {(const float*)d_in[P_AFC1_W], (const float*)d_in[P_CFC1_W]};
  const float* fc1b[2]   = {(const float*)d_in[P_AFC1_B], (const float*)d_in[P_CFC1_B]};
  const float* fc2w[2]   = {(const float*)d_in[P_AFC2_W], (const float*)d_in[P_CFC2_W]};
  const float* fc2b[2]   = {(const float*)d_in[P_AFC2_B], (const float*)d_in[P_CFC2_B]};
  float* out_s[2] = {(float*)d_out, (float*)d_out + (size_t)SEQA * HIDN};

  // mod-chunk offsets: [1s,1sc,1g, 2s,2sc,2g, 3s,3sc,3g] * 1536
  const int C1S = 0, C1SC = HIDN, C1G = 2 * HIDN;
  const int C2S = 3 * HIDN, C2SC = 4 * HIDN, C2G = 5 * HIDN;
  const int C3S = 6 * HIDN, C3SC = 7 * HIDN, C3G = 8 * HIDN;

  // 1) modulation vectors
  silu_vec_kernel<<<6, 256, 0, stream>>>(f_vec, sbuf);
  modvec_kernel<<<108, 256, 0, stream>>>(sbuf,
      (const float*)d_in[P_AMOD_W], (const float*)d_in[P_AMOD_B],
      (const float*)d_in[P_CMOD_W], (const float*)d_in[P_CMOD_B], amod, cmod);

  // 2) qkv for both streams -> rms-norm + rope -> packed f16 q/k/v
  for (int s = 0; s < 2; ++s) {
    ln_mod_kernel<<<SEQA, 256, 0, stream>>>(x_in[s], mods[s] + C1S, mods[s] + C1SC, xmod);
    launch_gemm(xmod, qkvw[s], big, SEQA, 3 * HIDN, HIDN, EPI_NONE, nullptr, nullptr,
                nullptr, stream);
    qkv_post_kernel<<<SEQA * NHEAD / 8, 256, 0, stream>>>(big, qn[s], kn[s], f_cos,
                                                          f_sin, qh, kh, vh, s);
  }

  // 3) joint self-attention over 4096 tokens
  flash_attn_kernel<<<dim3(SEQT / 128, NHEAD), 256, 0, stream>>>(qh, kh, vh, attno,
                                                                 SEQT, scale);

  // 4) self projection + gated residual
  for (int s = 0; s < 2; ++s)
    launch_gemm(attno + (size_t)s * SEQA * HIDN, sprojw[s], out_s[s], SEQA, HIDN,
                HIDN, EPI_GATE_RES, nullptr, mods[s] + C1G, x_in[s], stream);

  // 5) text kv + norms
  launch_gemm(f_text, (const float*)d_in[P_TEXT_KV_W], tkv, LTXT, 2 * HIDN, HIDN,
              EPI_NONE, nullptr, nullptr, nullptr, stream);
  rms_head_kernel<<<(LTXT * NHEAD + 7) / 8, 256, 0, stream>>>(
      tkv, 2 * HIDN, 0, (const float*)d_in[P_TEXT_KN], tkh, LTXT);
  rms_head_kernel<<<(LTXT * NHEAD + 7) / 8, 256, 0, stream>>>(
      tkv, 2 * HIDN, HIDN, nullptr, tvh, LTXT);

  // 6) cross attention per stream
  for (int s = 0; s < 2; ++s) {
    ln_mod_kernel<<<SEQA, 256, 0, stream>>>(out_s[s], mods[s] + C2S, mods[s] + C2SC, xmod);
    launch_gemm(xmod, cqw[s], big, SEQA, HIDN, HIDN, EPI_NONE, nullptr, nullptr,
                nullptr, stream);
    rms_head_kernel<<<SEQA * NHEAD / 8, 256, 0, stream>>>(big, HIDN, 0, cqn[s], qxh, SEQA);
    flash_attn_kernel<<<dim3(SEQA / 128, NHEAD), 256, 0, stream>>>(qxh, tkh, tvh,
                                                                   attno, LTXT, scale);
    launch_gemm(attno, cprojw[s], out_s[s], SEQA, HIDN, HIDN, EPI_GATE_RES, nullptr,
                mods[s] + C2G, out_s[s], stream);
  }

  // 7) MLP per stream
  for (int s = 0; s < 2; ++s) {
    ln_mod_kernel<<<SEQA, 256, 0, stream>>>(out_s[s], mods[s] + C3S, mods[s] + C3SC, xmod);
    launch_gemm(xmod, fc1w[s], big, SEQA, MLPH, HIDN, EPI_BIAS_GELU, fc1b[s], nullptr,
                nullptr, stream);
    launch_gemm(big, fc2w[s], out_s[s], SEQA, HIDN, MLPH, EPI_GATE_RES, fc2b[s],
                mods[s] + C3G, out_s[s], stream);
  }
}
